// Attention_7859790152297
// MI455X (gfx1250) — compile-verified, matrix operand order checked by reading, not run on hardware
//
#include <hip/hip_runtime.h>
#include <math.h>

// ---------------- problem constants ----------------
#define BB 8
#define NN_ 2049
#define CC 768
#define HH 12
#define DD 64
#define MM 256
#define LL 257          // landmarks incl. cls
#define LP 288          // padded to 9*32 for WMMA K-steps
#define KSZ 33
#define BH (BB*HH)      // 96
#define ROWS (BB*NN_)   // 16392

typedef unsigned short u16t;
typedef unsigned int   u32t;
typedef __attribute__((ext_vector_type(16))) __bf16          v16bf;
typedef __attribute__((ext_vector_type(8)))  float           v8f;
typedef __attribute__((ext_vector_type(8)))  unsigned short  us8;
typedef __attribute__((ext_vector_type(16))) unsigned short  us16;

// ---------------- helpers ----------------
__device__ __forceinline__ v8f vzero8() { v8f z = {}; return z; }

__device__ __forceinline__ u16t f2bf(float f) {          // RNE f32 -> bf16
  u32t u = __builtin_bit_cast(u32t, f);
  u32t r = u + 0x7FFFu + ((u >> 16) & 1u);
  return (u16t)(r >> 16);
}
__device__ __forceinline__ float bf2f(u16t h) {
  u32t u = ((u32t)h) << 16;
  return __builtin_bit_cast(float, u);
}

__device__ __forceinline__ v8f wmma_bf16(us16 a, us16 b, v8f c) {
  return __builtin_amdgcn_wmma_f32_16x16x32_bf16(
      false, __builtin_bit_cast(v16bf, a),
      false, __builtin_bit_cast(v16bf, b),
      (short)0, c, false, false);
}

// A/B(NT) operand: 16x32 tile, rows of a row-major (rows x ld) bf16 matrix.
// Lane L: row = L&15, half = L>>4; K chunks [h*8, h*8+7] and [16+h*8, 23+h*8].
__device__ __forceinline__ us16 load_op_rows(const u16t* p, int ld, int k0, int lane) {
  int r = lane & 15, hf = lane >> 4;
  const us8* c0 = (const us8*)(p + (size_t)r * ld + k0 + hf * 8);
  const us8* c1 = (const us8*)(p + (size_t)r * ld + k0 + 16 + hf * 8);
  us8 a = *c0, b = *c1;
  us16 o;
#pragma unroll
  for (int i = 0; i < 8; i++) { o[i] = a[i]; o[8 + i] = b[i]; }
  return o;
}

// B operand for NN GEMM: gather column n of a row-major (K x ldn) bf16 matrix.
__device__ __forceinline__ us16 load_op_cols(const u16t* p, int ldn, int n, int k0, int lane) {
  int hf = lane >> 4;
  us16 o;
#pragma unroll
  for (int j = 0; j < 8; j++) {
    o[j]     = p[(size_t)(k0 + hf * 8 + j) * ldn + n];
    o[8 + j] = p[(size_t)(k0 + 16 + hf * 8 + j) * ldn + n];
  }
  return o;
}

// ---------------- kernels ----------------
__global__ void k_cvt(const float* __restrict__ in, u16t* __restrict__ out, int n) {
  int i = blockIdx.x * 256 + threadIdx.x;
  if (i < n) out[i] = f2bf(in[i]);
}

// QKV GEMM: (16392x768) @ (2304x768)^T, bf16 WMMA, epilogue scatters Q*8,K*8,V
__global__ void k_qkv(const u16t* __restrict__ xb, const u16t* __restrict__ wb,
                      u16t* __restrict__ Qb, u16t* __restrict__ Kb, u16t* __restrict__ Vb) {
  __shared__ u16t As[128 * 32];
  __shared__ u16t Bs[128 * 32];
  int tid = threadIdx.x;
  int lane = tid & 31, wv = tid >> 5;
  int row0 = blockIdx.x * 128;
  int col0 = blockIdx.y * 128;
  v8f acc[8];
#pragma unroll
  for (int f = 0; f < 8; f++) acc[f] = vzero8();

  for (int kk = 0; kk < 768; kk += 32) {
    {
      int r = tid >> 1, kc = (tid & 1) * 16;
      int gr = row0 + r;
      us8 z = {};
      const us8* sx = (const us8*)(xb + (size_t)gr * 768 + kk + kc);
      ((us8*)(As + r * 32 + kc))[0] = (gr < ROWS) ? sx[0] : z;
      ((us8*)(As + r * 32 + kc))[1] = (gr < ROWS) ? sx[1] : z;
      const us8* sw = (const us8*)(wb + (size_t)(col0 + r) * 768 + kk + kc);
      ((us8*)(Bs + r * 32 + kc))[0] = sw[0];
      ((us8*)(Bs + r * 32 + kc))[1] = sw[1];
      __builtin_prefetch(xb + (size_t)gr * 768 + kk + 32, 0, 3);
    }
    __syncthreads();
    us16 a = load_op_rows(As + wv * 16 * 32, 32, 0, lane);
#pragma unroll
    for (int f = 0; f < 8; f++) {
      us16 b = load_op_rows(Bs + f * 16 * 32, 32, 0, lane);
      acc[f] = wmma_bf16(a, b, acc[f]);
    }
    __syncthreads();
  }
#pragma unroll
  for (int f = 0; f < 8; f++) {
    int c = col0 + f * 16 + (lane & 15);
    int g = c / 768, rem = c % 768;
    int h = rem >> 6, d = rem & 63;
    float s = (g < 2) ? 8.0f : 1.0f;  // Q,K divided by SCALE=D^-0.5=0.125
    u16t* dst = (g == 0) ? Qb : (g == 1) ? Kb : Vb;
#pragma unroll
    for (int v = 0; v < 8; v++) {
      int m = row0 + wv * 16 + (lane >> 4) * 8 + v;
      if (m < ROWS) {
        int b = m / NN_, n = m % NN_;
        dst[(((size_t)b * HH + h) * NN_ + n) * DD + d] = f2bf(acc[f][v] * s);
      }
    }
  }
}

// landmarks: cls + 256 means over groups of 8
__global__ void k_land(const u16t* __restrict__ Qb, const u16t* __restrict__ Kb,
                       u16t* __restrict__ Qlb, u16t* __restrict__ Klb) {
  int i = blockIdx.x * 256 + threadIdx.x;
  if (i >= BH * LP * DD) return;
  int d = i & 63;
  int m = (i >> 6) % LP;
  int bh = i / (LP * DD);
  const u16t* q = Qb + (size_t)bh * NN_ * DD;
  const u16t* k = Kb + (size_t)bh * NN_ * DD;
  float qv = 0.f, kv = 0.f;
  if (m == 0) { qv = bf2f(q[d]); kv = bf2f(k[d]); }
  else if (m <= MM) {
    int n0 = 1 + (m - 1) * 8;
    for (int t = 0; t < 8; t++) {
      qv += bf2f(q[(size_t)(n0 + t) * DD + d]);
      kv += bf2f(k[(size_t)(n0 + t) * DD + d]);
    }
    qv *= 0.125f; kv *= 0.125f;
  }
  Qlb[i] = f2bf(qv); Klb[i] = f2bf(kv);
}

// kernel_2 = softmax(Ql @ Kl^T), 16 rows per wave, S tile in LDS
__global__ void k_k2soft(const u16t* __restrict__ Qlb, const u16t* __restrict__ Klb,
                         u16t* __restrict__ K2b) {
  __shared__ float S[16 * LP];
  __shared__ float rm[16], rs[16];
  int lane = threadIdx.x;
  int bh = blockIdx.y;
  int row0 = blockIdx.x * 16;
  const u16t* qa = Qlb + (size_t)bh * LP * DD + (size_t)row0 * DD;
  for (int nt = 0; nt < LP / 16; nt++) {
    v8f acc = vzero8();
    const u16t* kb = Klb + (size_t)bh * LP * DD + (size_t)nt * 16 * DD;
    for (int ks = 0; ks < DD; ks += 32) {
      us16 a = load_op_rows(qa, DD, ks, lane);
      us16 b = load_op_rows(kb, DD, ks, lane);
      acc = wmma_bf16(a, b, acc);
    }
    int c = nt * 16 + (lane & 15);
#pragma unroll
    for (int v = 0; v < 8; v++) S[((lane >> 4) * 8 + v) * LP + c] = acc[v];
  }
  __syncthreads();
  if (lane < 16) {
    int gr = row0 + lane;
    float mx = -3.0e38f;
    for (int c = 0; c < LL; c++) mx = fmaxf(mx, S[lane * LP + c]);
    float s = 0.f;
    for (int c = 0; c < LL; c++) s += __expf(S[lane * LP + c] - mx);
    rm[lane] = mx;
    rs[lane] = (gr < LL) ? 1.0f / s : 0.0f;  // pad rows -> zero
  }
  __syncthreads();
  for (int idx = lane; idx < 16 * LP; idx += 32) {
    int r = idx / LP, c = idx % LP;
    float val = (c < LL && rs[r] != 0.0f) ? __expf(S[idx] - rm[r]) * rs[r] : 0.0f;
    int gr = row0 + r;
    K2b[((size_t)bh * LP + gr) * LP + c] = f2bf(val);
  }
}

__global__ void k_scal0(u32t* scal) { if (threadIdx.x == 0) scal[0] = 0u; }

// max column-sum of |K2| (rows sum to 1 exactly: softmax)
__global__ void k_colmax(const u16t* __restrict__ K2b, u32t* scal) {
  int c = threadIdx.x, bh = blockIdx.x;
  if (c >= LL) return;
  float s = 0.f;
  for (int r = 0; r < LL; r++) s += bf2f(K2b[((size_t)bh * LP + r) * LP + c]);
  atomicMax(scal, __builtin_bit_cast(u32t, s));  // positive floats: int-order ok
}

// V0 = K2^T / (max_colsum * max_rowsum), max_rowsum == 1
__global__ void k_v0(const u16t* __restrict__ K2b, const u32t* __restrict__ scal,
                     u16t* __restrict__ V0) {
  int i = blockIdx.x * 256 + threadIdx.x;
  if (i >= BH * LP * LP) return;
  int j = i % LP;
  int r = (i / LP) % LP;
  int bh = i / (LP * LP);
  float inv = 1.0f / __builtin_bit_cast(float, scal[0]);
  V0[i] = f2bf(bf2f(K2b[((size_t)bh * LP + j) * LP + r]) * inv);
}

// batched NN GEMM (288 x 288) @ (288 x Nc), epilogue out = m*P + d*I(diag<257)
__global__ void k_gemm(const u16t* __restrict__ A, const u16t* __restrict__ Bm,
                       u16t* __restrict__ out1, u16t* __restrict__ out2,
                       int Nc, float d1, float m1, float d2, float m2) {
  int lane = threadIdx.x;
  int bh = blockIdx.z;
  int row0 = blockIdx.x * 16;
  int c0 = blockIdx.y * 64;
  const u16t* Ab = A + (size_t)bh * LP * LP + (size_t)row0 * LP;
  const u16t* Bb = Bm + (size_t)bh * LP * Nc;
  v8f acc[4];
#pragma unroll
  for (int t = 0; t < 4; t++) acc[t] = vzero8();
  int ntiles = (Nc - c0 + 15) / 16; if (ntiles > 4) ntiles = 4;
  for (int ks = 0; ks < LP; ks += 32) {
    us16 a = load_op_rows(Ab, LP, ks, lane);
    __builtin_prefetch(Ab + (lane & 15) * LP + ks + 32, 0, 3);
    for (int t = 0; t < ntiles; t++) {
      us16 b = load_op_cols(Bb, Nc, c0 + t * 16 + (lane & 15), ks, lane);
      acc[t] = wmma_bf16(a, b, acc[t]);
    }
  }
  for (int t = 0; t < ntiles; t++) {
    int c = c0 + t * 16 + (lane & 15);
#pragma unroll
    for (int v = 0; v < 8; v++) {
      int gr = row0 + (lane >> 4) * 8 + v;
      float diag = (gr == c && gr < LL) ? 1.0f : 0.0f;
      size_t off = ((size_t)bh * LP + gr) * Nc + c;
      out1[off] = f2bf(m1 * acc[t][v] + d1 * diag);
      if (out2) out2[off] = f2bf(m2 * acc[t][v] + d2 * diag);
    }
  }
}

// K3V = softmax(Ql @ K^T, over N=2049) @ V   — two-pass flash per 16-row wave
__global__ void k_k3v(const u16t* __restrict__ Qlb, const u16t* __restrict__ Kb,
                      const u16t* __restrict__ Vb, u16t* __restrict__ K3Vb) {
  __shared__ u16t Pl[16 * 32];
  __shared__ u16t Vs[32 * 64];
  __shared__ float rowm[16], rowsum[16];
  int lane = threadIdx.x;
  int bh = blockIdx.y;
  int row0 = blockIdx.x * 16;
  const u16t* qa = Qlb + (size_t)bh * LP * DD + (size_t)row0 * DD;
  const u16t* kbase = Kb + (size_t)bh * NN_ * DD;
  const u16t* vbase = Vb + (size_t)bh * NN_ * DD;

  float mloc[8];
#pragma unroll
  for (int v = 0; v < 8; v++) mloc[v] = -3.0e38f;
  // pass 1: per-row max
  for (int c0 = 0; c0 < NN_; c0 += 32) {
#pragma unroll
    for (int t = 0; t < 2; t++) {
      v8f acc = vzero8();
      for (int ks = 0; ks < DD; ks += 32) {
        us16 a = load_op_rows(qa, DD, ks, lane);
        us16 b = load_op_rows(kbase + (size_t)(c0 + t * 16) * DD, DD, ks, lane);
        acc = wmma_bf16(a, b, acc);
      }
      int col = c0 + t * 16 + (lane & 15);
      if (col < NN_) {
#pragma unroll
        for (int v = 0; v < 8; v++) mloc[v] = fmaxf(mloc[v], acc[v]);
      }
    }
  }
#pragma unroll
  for (int v = 0; v < 8; v++)
    for (int m = 1; m < 16; m <<= 1) mloc[v] = fmaxf(mloc[v], __shfl_xor(mloc[v], m, 32));
  if (lane == 0 || lane == 16) {
#pragma unroll
    for (int v = 0; v < 8; v++) rowm[(lane >> 4) * 8 + v] = mloc[v];
  }
  __syncthreads();
  float mr[8];
#pragma unroll
  for (int v = 0; v < 8; v++) mr[v] = rowm[(lane >> 4) * 8 + v];

  // pass 2: exp + PV accumulation
  float sloc[8];
#pragma unroll
  for (int v = 0; v < 8; v++) sloc[v] = 0.f;
  v8f accO[4];
#pragma unroll
  for (int t = 0; t < 4; t++) accO[t] = vzero8();
  for (int c0 = 0; c0 < NN_; c0 += 32) {
    {
      const us8* src = (const us8*)(vbase + (size_t)(c0 + lane) * DD);
#pragma unroll
      for (int j = 0; j < 8; j++) ((us8*)(Vs + lane * 64))[j] = src[j];
    }
#pragma unroll
    for (int t = 0; t < 2; t++) {
      v8f acc = vzero8();
      for (int ks = 0; ks < DD; ks += 32) {
        us16 a = load_op_rows(qa, DD, ks, lane);
        us16 b = load_op_rows(kbase + (size_t)(c0 + t * 16) * DD, DD, ks, lane);
        acc = wmma_bf16(a, b, acc);
      }
      int col = c0 + t * 16 + (lane & 15);
      int cl = t * 16 + (lane & 15);
#pragma unroll
      for (int v = 0; v < 8; v++) {
        float p = (col < NN_) ? __expf(acc[v] - mr[v]) : 0.0f;
        sloc[v] += p;
        Pl[((lane >> 4) * 8 + v) * 32 + cl] = f2bf(p);
      }
    }
    __syncthreads();
    us16 pa = load_op_rows(Pl, 32, 0, lane);
#pragma unroll
    for (int t = 0; t < 4; t++) {
      us16 b = load_op_cols(Vs, 64, t * 16 + (lane & 15), 0, lane);
      accO[t] = wmma_bf16(pa, b, accO[t]);
    }
    __syncthreads();
  }
#pragma unroll
  for (int v = 0; v < 8; v++)
    for (int m = 1; m < 16; m <<= 1) sloc[v] += __shfl_xor(sloc[v], m, 32);
  if (lane == 0 || lane == 16) {
#pragma unroll
    for (int v = 0; v < 8; v++) rowsum[(lane >> 4) * 8 + v] = sloc[v];
  }
  __syncthreads();
#pragma unroll
  for (int t = 0; t < 4; t++) {
    int d = t * 16 + (lane & 15);
#pragma unroll
    for (int v = 0; v < 8; v++) {
      int r = (lane >> 4) * 8 + v;
      int gr = row0 + r;
      float val = (gr < LL) ? accO[t][v] / rowsum[r] : 0.0f;
      K3Vb[((size_t)bh * LP + gr) * DD + d] = f2bf(val);
    }
  }
}

// out = softmax(Q @ Kl^T) @ Pmat + depthwise_conv(V), write (B,N,C) f32
__global__ void k_out(const u16t* __restrict__ Qb, const u16t* __restrict__ Klb,
                      const u16t* __restrict__ Pmatb, const u16t* __restrict__ Vb,
                      const float* __restrict__ wconv, float* __restrict__ out) {
  __shared__ float S[16 * LP];
  __shared__ u16t Pl[16 * LP];
  __shared__ float Vc[48 * 64];
  __shared__ float rm[16], rs[16];
  int lane = threadIdx.x;
  int bh = blockIdx.y;
  int b = bh / HH, h = bh % HH;
  int n0 = blockIdx.x * 16;
  const u16t* qa = Qb + (size_t)bh * NN_ * DD + (size_t)n0 * DD;
  const u16t* klb = Klb + (size_t)bh * LP * DD;

  for (int nt = 0; nt < LP / 16; nt++) {
    v8f acc = vzero8();
    for (int ks = 0; ks < DD; ks += 32) {
      us16 a = load_op_rows(qa, DD, ks, lane);
      us16 bb = load_op_rows(klb + (size_t)nt * 16 * DD, DD, ks, lane);
      acc = wmma_bf16(a, bb, acc);
    }
    int c = nt * 16 + (lane & 15);
#pragma unroll
    for (int v = 0; v < 8; v++) S[((lane >> 4) * 8 + v) * LP + c] = acc[v];
  }
  __syncthreads();
  if (lane < 16) {
    float mx = -3.0e38f;
    for (int c = 0; c < LL; c++) mx = fmaxf(mx, S[lane * LP + c]);
    float s = 0.f;
    for (int c = 0; c < LL; c++) s += __expf(S[lane * LP + c] - mx);
    rm[lane] = mx; rs[lane] = 1.0f / s;
  }
  __syncthreads();
  for (int idx = lane; idx < 16 * LP; idx += 32) {
    int r = idx / LP, c = idx % LP;
    Pl[idx] = f2bf((c < LL) ? __expf(S[idx] - rm[r]) * rs[r] : 0.0f);
  }
  for (int li = lane; li < 48 * 64; li += 32) {  // V halo for conv
    int j = li >> 6, d = li & 63;
    int n = n0 - 16 + j;
    Vc[li] = (n >= 0 && n < NN_) ? bf2f(Vb[(size_t)bh * NN_ * DD + (size_t)n * DD + d]) : 0.0f;
  }
  __syncthreads();

  v8f accO[4];
#pragma unroll
  for (int t = 0; t < 4; t++) accO[t] = vzero8();
  const u16t* pm = Pmatb + (size_t)bh * LP * DD;
  for (int ks = 0; ks < LP; ks += 32) {
    us16 a = load_op_rows(Pl, LP, ks, lane);
#pragma unroll
    for (int t = 0; t < 4; t++) {
      us16 bb = load_op_cols(pm, DD, t * 16 + (lane & 15), ks, lane);
      accO[t] = wmma_bf16(a, bb, accO[t]);
    }
  }
#pragma unroll
  for (int t = 0; t < 4; t++) {
    int d = t * 16 + (lane & 15);
#pragma unroll
    for (int v = 0; v < 8; v++) {
      int r = (lane >> 4) * 8 + v;
      int n = n0 + r;
      if (n < NN_) {
        float cv = 0.0f;
        for (int k = 0; k < KSZ; k++) cv += Vc[(r + k) * 64 + d] * wconv[h * KSZ + k];
        out[((size_t)b * NN_ + n) * CC + h * DD + d] = accO[t][v] + cv;
      }
    }
  }
}

// ---------------- host launcher ----------------
extern "C" void kernel_launch(void* const* d_in, const int* in_sizes, int n_in,
                              void* d_out, int out_size, void* d_ws, size_t ws_size,
                              hipStream_t stream) {
  (void)in_sizes; (void)n_in; (void)out_size; (void)ws_size;
  const float* x     = (const float*)d_in[0];
  const float* wqkv  = (const float*)d_in[1];
  const float* wconv = (const float*)d_in[2];
  float* out = (float*)d_out;

  char* ws = (char*)d_ws;
  size_t off = 0;
  auto alloc = [&](size_t bytes) -> void* {
    void* p = ws + off;
    off += (bytes + 255) & ~(size_t)255;
    return p;
  };
  u16t* xb   = (u16t*)alloc((size_t)ROWS * CC * 2);
  u16t* wb   = (u16t*)alloc((size_t)3 * CC * CC * 2);
  u16t* Qb   = (u16t*)alloc((size_t)BH * NN_ * DD * 2);
  u16t* Kb   = (u16t*)alloc((size_t)BH * NN_ * DD * 2);
  u16t* Vb   = (u16t*)alloc((size_t)BH * NN_ * DD * 2);
  u16t* Qlb  = (u16t*)alloc((size_t)BH * LP * DD * 2);
  u16t* Klb  = (u16t*)alloc((size_t)BH * LP * DD * 2);
  u16t* K2b  = (u16t*)alloc((size_t)BH * LP * LP * 2);
  u16t* Vit0 = (u16t*)alloc((size_t)BH * LP * LP * 2);
  u16t* Vit1 = (u16t*)alloc((size_t)BH * LP * LP * 2);
  u16t* Amat = (u16t*)alloc((size_t)BH * LP * LP * 2);
  u16t* T1   = (u16t*)alloc((size_t)BH * LP * LP * 2);
  u16t* T2   = (u16t*)alloc((size_t)BH * LP * LP * 2);
  u16t* K3Vb = (u16t*)alloc((size_t)BH * LP * DD * 2);
  u16t* Pmb  = (u16t*)alloc((size_t)BH * LP * DD * 2);
  u32t* scal = (u32t*)alloc(256);

  k_cvt<<<(ROWS * CC + 255) / 256, 256, 0, stream>>>(x, xb, ROWS * CC);
  k_cvt<<<(3 * CC * CC + 255) / 256, 256, 0, stream>>>(wqkv, wb, 3 * CC * CC);
  k_qkv<<<dim3((ROWS + 127) / 128, (3 * CC) / 128), 256, 0, stream>>>(xb, wb, Qb, Kb, Vb);
  k_land<<<(BH * LP * DD + 255) / 256, 256, 0, stream>>>(Qb, Kb, Qlb, Klb);
  k_k2soft<<<dim3(LP / 16, BH), 32, 0, stream>>>(Qlb, Klb, K2b);
  k_scal0<<<1, 32, 0, stream>>>(scal);
  k_colmax<<<BH, LP, 0, stream>>>(K2b, scal);
  k_v0<<<(BH * LP * LP + 255) / 256, 256, 0, stream>>>(K2b, scal, Vit0);

  u16t* Vc = Vit0; u16t* Vn = Vit1;
  for (int it = 0; it < 6; it++) {
    // A = K2 @ V ; T1 = 7I - A
    k_gemm<<<dim3(LP / 16, 5, BH), 32, 0, stream>>>(K2b, Vc, Amat, T1, LP, 0.f, 1.f, 7.f, -1.f);
    // T2 = 15I - A @ T1
    k_gemm<<<dim3(LP / 16, 5, BH), 32, 0, stream>>>(Amat, T1, T2, nullptr, LP, 15.f, -1.f, 0.f, 0.f);
    // T1 = 13I - A @ T2
    k_gemm<<<dim3(LP / 16, 5, BH), 32, 0, stream>>>(Amat, T2, T1, nullptr, LP, 13.f, -1.f, 0.f, 0.f);
    // Vn = 0.25 * V @ T1
    k_gemm<<<dim3(LP / 16, 5, BH), 32, 0, stream>>>(Vc, T1, Vn, nullptr, LP, 0.f, 0.25f, 0.f, 0.f);
    u16t* tmp = Vc; Vc = Vn; Vn = tmp;
  }

  k_k3v<<<dim3(LP / 16, BH), 32, 0, stream>>>(Qlb, Kb, Vb, K3Vb);
  // Pmat = inv(kernel_2) @ (kernel_3 @ V)
  k_gemm<<<dim3(LP / 16, 1, BH), 32, 0, stream>>>(Vc, K3Vb, Pmb, nullptr, DD, 0.f, 1.f, 0.f, 0.f);
  k_out<<<dim3((NN_ + 15) / 16, BH), 32, 0, stream>>>(Qb, Klb, Pmb, Vb, wconv, out);
}
